// AttnHGCN_1941325218035
// MI455X (gfx1250) — matrix operand order repaired and checked by reading, not compiled, not run on hardware
//
#include <hip/hip_runtime.h>
#include <hip/hip_bf16.h>
#include <math.h>

typedef __attribute__((ext_vector_type(16))) __bf16 v16bf;
typedef __attribute__((ext_vector_type(8)))  float  v8f;

#define CHANNEL 64
#define INV_SQRT_DK 0.17677669529663687f /* 1/sqrt(32) */

// ---------------------------------------------------------------------------
// Ordered-uint encoding of float for atomicMax-based segment max.
// ---------------------------------------------------------------------------
__device__ __forceinline__ unsigned int fkey(float f) {
    unsigned int u = __float_as_uint(f);
    return (u & 0x80000000u) ? ~u : (u | 0x80000000u);
}
__device__ __forceinline__ float fval(unsigned int k) {
    unsigned int u = (k & 0x80000000u) ? (k & 0x7FFFFFFFu) : ~k;
    return __uint_as_float(u);
}

// ---------------------------------------------------------------------------
// Pack two contiguous 8-float runs into one v16bf operand.
// ---------------------------------------------------------------------------
__device__ __forceinline__ v16bf pack_bf16(float4 x0, float4 x1, float4 y0, float4 y1) {
    v16bf a;
    a[0]  = (__bf16)x0.x; a[1]  = (__bf16)x0.y; a[2]  = (__bf16)x0.z; a[3]  = (__bf16)x0.w;
    a[4]  = (__bf16)x1.x; a[5]  = (__bf16)x1.y; a[6]  = (__bf16)x1.z; a[7]  = (__bf16)x1.w;
    a[8]  = (__bf16)y0.x; a[9]  = (__bf16)y0.y; a[10] = (__bf16)y0.z; a[11] = (__bf16)y0.w;
    a[12] = (__bf16)y1.x; a[13] = (__bf16)y1.y; a[14] = (__bf16)y1.z; a[15] = (__bf16)y1.w;
    return a;
}

// B 32x16 operand (CDNA5 7.12.2): lane L -> col N = nb + (L&15);
// element e -> K = kb + (e&7) + ((e>>3)<<4) + ((L>>4)<<3).
__device__ __forceinline__ v16bf make_B_tile(const float* __restrict__ W,
                                             int lane, int kb, int nb) {
    v16bf b;
    int n = nb + (lane & 15);
    int koff = kb + ((lane >> 4) << 3);
#pragma unroll
    for (int e = 0; e < 16; ++e) {
        b[e] = (__bf16)W[(size_t)(koff + (e & 7) + ((e >> 3) << 4)) * CHANNEL + n];
    }
    return b;
}

// ---------------------------------------------------------------------------
// Kernel 1: Q = ent_cur @ W_Q  (M=n_ent, N=64, K=64) via v_wmma_f32_16x16x32_bf16.
// Block = 256 threads = 8 waves; each wave computes a 16x64 output tile.
//  - B tiles staged once per block in LDS (per-lane WMMA layout, 8 KB)
//  - A loaded as float4 pairs (the A layout's K pattern is 2 runs of 8 floats)
//  - D staged through LDS -> 8 coalesced global_store_b128 per lane
// ---------------------------------------------------------------------------
__global__ void __launch_bounds__(256)
qmat_kernel(const float* __restrict__ ent, const float* __restrict__ W,
            float* __restrict__ Q, int n_rows) {
    __shared__ __align__(32) __bf16 sB[8][32][16];     // [kb2*4+nt][lane][elem]
    __shared__ __align__(16) float  sOut[8][16 * CHANNEL];

    int tid  = threadIdx.x;
    int lane = tid & 31;
    int wIdx = tid >> 5;                               // wave in block

    // ---- Stage all 8 B tiles (kb in {0,32} x nt in {0..3}) cooperatively ----
    {
        int tileId  = wIdx;                            // 8 waves -> 8 tiles
        int kb      = (tileId >> 2) * 32;
        int nb      = (tileId & 3) * 16;
        v16bf b = make_B_tile(W, lane, kb, nb);
        *(v16bf*)&sB[tileId][lane][0] = b;
    }
    __syncthreads();

    int wave = (blockIdx.x * blockDim.x + tid) >> 5;
    int row0 = wave * 16;
    if (row0 >= n_rows) return;
    int rows_avail = n_rows - row0;
    if (rows_avail > 16) rows_avail = 16;

    // ---- Load A operands (16x32 bf16 per K-chunk) ----
    // A layout: lane L -> row M = L&15; K = kb + (e&7) + ((e>>3)<<4) + ((L>>4)<<3)
    int m = lane & 15;
    if (m >= rows_avail) m = rows_avail - 1;           // tail clamp (read only)
    const float4* row4 = (const float4*)(ent + ((size_t)row0 + m) * CHANNEL);
    int h2 = (lane >> 4) * 2;                          // float4 offset of K-run
    float4 a00 = row4[h2 + 0], a01 = row4[h2 + 1];     // K kb+0..7
    float4 a02 = row4[h2 + 4], a03 = row4[h2 + 5];     // K kb+16..23
    float4 a10 = row4[h2 + 8], a11 = row4[h2 + 9];     // kb=32 chunk
    float4 a12 = row4[h2 + 12], a13 = row4[h2 + 13];
    v16bf a0 = pack_bf16(a00, a01, a02, a03);
    v16bf a1 = pack_bf16(a10, a11, a12, a13);

    // ---- 4 N-tiles x 2 chained WMMAs ----
    int n_lane = lane & 15;
    int mbase  = (lane >> 4) * 8;
    float* outw = &sOut[wIdx][0];
#pragma unroll
    for (int nt = 0; nt < 4; ++nt) {
        v16bf b0 = *(const v16bf*)&sB[nt][lane][0];
        v16bf b1 = *(const v16bf*)&sB[4 + nt][lane][0];
        v8f c = {};
        c = __builtin_amdgcn_wmma_f32_16x16x32_bf16(false, a0, false, b0,
                                                    (short)0, c, false, false);
        c = __builtin_amdgcn_wmma_f32_16x16x32_bf16(false, a1, false, b1,
                                                    (short)0, c, false, false);
        int n = nt * 16 + n_lane;
#pragma unroll
        for (int r = 0; r < 8; ++r)
            outw[(mbase + r) * CHANNEL + n] = c[r];    // LDS, conflict-free
    }

    // ---- Coalesced writeback: 8 x global_store_b128 per lane ----
    float* qbase = Q + (size_t)row0 * CHANNEL;
#pragma unroll
    for (int j = 0; j < 8; ++j) {
        int flat = lane * 4 + j * 128;                 // 512B contiguous / store
        int r    = flat >> 6;
        if (r < rows_avail) {
            float4 v = *(const float4*)&outw[flat];
            *(float4*)&qbase[flat] = v;
        }
    }
}

// ---------------------------------------------------------------------------
// Kernel 2: per-edge scores + atomic segment max.
// score[e,h] = sum_k Q[head][h*32+k] * (Q[tail][h*32+k] * rel[h*32+k]) / sqrt(Dk)
// ---------------------------------------------------------------------------
__global__ void __launch_bounds__(256)
score_kernel(const float4* __restrict__ Q4, const int* __restrict__ head,
             const int* __restrict__ tail, const int* __restrict__ etype,
             const float4* __restrict__ rel4, float* __restrict__ score,
             unsigned int* __restrict__ segmax, int E) {
    int e = blockIdx.x * blockDim.x + threadIdx.x;
    if (e >= E) return;
    int h = head[e], t = tail[e], r = etype[e] - 1;
    const float4* qh = Q4 + (size_t)h * 16;
    const float4* qt = Q4 + (size_t)t * 16;
    const float4* rl = rel4 + (size_t)r * 16;
    float s0 = 0.f, s1 = 0.f;
#pragma unroll
    for (int i = 0; i < 8; ++i) {
        float4 a = qh[i], b = qt[i], c = rl[i];
        s0 += a.x * b.x * c.x + a.y * b.y * c.y + a.z * b.z * c.z + a.w * b.w * c.w;
    }
#pragma unroll
    for (int i = 8; i < 16; ++i) {
        float4 a = qh[i], b = qt[i], c = rl[i];
        s1 += a.x * b.x * c.x + a.y * b.y * c.y + a.z * b.z * c.z + a.w * b.w * c.w;
    }
    s0 *= INV_SQRT_DK;
    s1 *= INV_SQRT_DK;
    score[2 * e]     = s0;
    score[2 * e + 1] = s1;
    atomicMax(&segmax[2 * h],     fkey(s0));
    atomicMax(&segmax[2 * h + 1], fkey(s1));
}

// ---------------------------------------------------------------------------
// Kernel 3: e = exp(score - segmax); atomic segment sum. One thread per (edge,head).
// ---------------------------------------------------------------------------
__global__ void __launch_bounds__(256)
expsum_kernel(float* __restrict__ score, const int* __restrict__ head,
              const unsigned int* __restrict__ segmax, float* __restrict__ segsum,
              int E) {
    int idx = blockIdx.x * blockDim.x + threadIdx.x;
    if (idx >= 2 * E) return;
    int e  = idx >> 1;
    int hd = idx & 1;
    int h  = head[e];
    float m  = fval(segmax[2 * h + hd]);
    float ex = expf(score[idx] - m);
    score[idx] = ex;
    atomicAdd(&segsum[2 * h + hd], ex);
}

// ---------------------------------------------------------------------------
// Kernel 4: attention-weighted scatter: ent_agg[head] += (ent_cur[tail]*rel)*attn
// One thread per (edge, group-of-4-channels) => 16 threads/edge.
// ---------------------------------------------------------------------------
__global__ void __launch_bounds__(256)
agg_kernel(const float* __restrict__ escore, const float* __restrict__ segsum,
           const int* __restrict__ head, const int* __restrict__ tail,
           const int* __restrict__ etype, const float4* __restrict__ rel4,
           const float4* __restrict__ ent4, float* __restrict__ ent_agg,
           long long nwork) {
    long long idx = (long long)blockIdx.x * blockDim.x + threadIdx.x;
    if (idx >= nwork) return;
    int e = (int)(idx >> 4);
    int g = (int)(idx & 15);
    int h = head[e], t = tail[e], r = etype[e] - 1;
    int hd = g >> 3;
    float attn = escore[2 * e + hd] / segsum[2 * h + hd];
    float4 v  = ent4[(size_t)t * 16 + g];
    float4 rw = rel4[(size_t)r * 16 + g];
    float* dst = ent_agg + (size_t)h * CHANNEL + g * 4;
    atomicAdd(dst + 0, v.x * rw.x * attn);
    atomicAdd(dst + 1, v.y * rw.y * attn);
    atomicAdd(dst + 2, v.z * rw.z * attn);
    atomicAdd(dst + 3, v.w * rw.w * attn);
}

// ---------------------------------------------------------------------------
// Kernel 5: user scatter: usr_agg[u] += w * ent_cur[i]. 16 threads/inter-edge.
// ---------------------------------------------------------------------------
__global__ void __launch_bounds__(256)
user_agg_kernel(const float* __restrict__ iw, const int* __restrict__ u_idx,
                const int* __restrict__ i_idx, const float4* __restrict__ ent4,
                float* __restrict__ usr_agg, long long nwork) {
    long long idx = (long long)blockIdx.x * blockDim.x + threadIdx.x;
    if (idx >= nwork) return;
    int e = (int)(idx >> 4);
    int g = (int)(idx & 15);
    int u = u_idx[e], i = i_idx[e];
    float w = iw[e];
    float4 v = ent4[(size_t)i * 16 + g];
    float* dst = usr_agg + (size_t)u * CHANNEL + g * 4;
    atomicAdd(dst + 0, v.x * w);
    atomicAdd(dst + 1, v.y * w);
    atomicAdd(dst + 2, v.z * w);
    atomicAdd(dst + 3, v.w * w);
}

// ---------------------------------------------------------------------------
// Kernel 6: cur = l2norm(agg); res += cur. One thread per row (64 channels).
// ---------------------------------------------------------------------------
__global__ void __launch_bounds__(256)
norm_res_kernel(const float4* __restrict__ agg, float4* __restrict__ cur,
                float4* __restrict__ res, int n_rows) {
    int row = blockIdx.x * blockDim.x + threadIdx.x;
    if (row >= n_rows) return;
    const float4* a = agg + (size_t)row * 16;
    float4 v[16];
    float ss = 0.f;
#pragma unroll
    for (int i = 0; i < 16; ++i) {
        v[i] = a[i];
        ss += v[i].x * v[i].x + v[i].y * v[i].y + v[i].z * v[i].z + v[i].w * v[i].w;
    }
    float scale = 1.0f / fmaxf(sqrtf(ss), 1e-12f);
    float4* c = cur + (size_t)row * 16;
    float4* o = res + (size_t)row * 16;
#pragma unroll
    for (int i = 0; i < 16; ++i) {
        float4 nv = make_float4(v[i].x * scale, v[i].y * scale,
                                v[i].z * scale, v[i].w * scale);
        c[i] = nv;
        float4 ov = o[i];
        o[i] = make_float4(ov.x + nv.x, ov.y + nv.y, ov.z + nv.z, ov.w + nv.w);
    }
}

// ---------------------------------------------------------------------------
// Host orchestration (graph-capture safe: only async ops on `stream`).
// ---------------------------------------------------------------------------
extern "C" void kernel_launch(void* const* d_in, const int* in_sizes, int n_in,
                              void* d_out, int out_size, void* d_ws, size_t ws_size,
                              hipStream_t stream) {
    const float* user_emb     = (const float*)d_in[0];
    const float* entity_emb   = (const float*)d_in[1];
    const float* relation_emb = (const float*)d_in[2];
    const float* W_Q          = (const float*)d_in[3];
    const float* inter_w      = (const float*)d_in[4];
    const int*   edge_index   = (const int*)d_in[5];
    const int*   edge_type    = (const int*)d_in[6];
    const int*   inter_edge   = (const int*)d_in[7];

    const int C       = CHANNEL;
    const int n_users = in_sizes[0] / C;
    const int n_ent   = in_sizes[1] / C;
    const int E       = in_sizes[6];
    const int NI      = in_sizes[4];

    const int* head  = edge_index;
    const int* tail  = edge_index + E;
    const int* u_idx = inter_edge;
    const int* i_idx = inter_edge + NI;

    float* ws = (float*)d_ws;
    size_t off = 0;
    float*        Q       = ws + off;                       off += (size_t)n_ent * C;
    float*        score   = ws + off;                       off += (size_t)E * 2;
    unsigned int* segmax  = (unsigned int*)(ws + off);      off += (size_t)n_ent * 2;
    float*        segsum  = ws + off;                       off += (size_t)n_ent * 2;
    float*        ent_agg = ws + off;                       off += (size_t)n_ent * C;
    float*        usr_agg = ws + off;                       off += (size_t)n_users * C;
    float*        ent_cur = ws + off;                       off += (size_t)n_ent * C;
    float*        usr_cur = ws + off;                       off += (size_t)n_users * C;

    float* out_ent = (float*)d_out;
    float* out_usr = out_ent + (size_t)n_ent * C;

    // Initialize residuals and current embeddings.
    hipMemcpyAsync(ent_cur, entity_emb, (size_t)n_ent * C * sizeof(float),
                   hipMemcpyDeviceToDevice, stream);
    hipMemcpyAsync(usr_cur, user_emb, (size_t)n_users * C * sizeof(float),
                   hipMemcpyDeviceToDevice, stream);
    hipMemcpyAsync(out_ent, entity_emb, (size_t)n_ent * C * sizeof(float),
                   hipMemcpyDeviceToDevice, stream);
    hipMemcpyAsync(out_usr, user_emb, (size_t)n_users * C * sizeof(float),
                   hipMemcpyDeviceToDevice, stream);

    for (int hop = 0; hop < 2; ++hop) {
        hipMemsetAsync(segmax,  0, (size_t)n_ent * 2 * sizeof(unsigned int), stream);
        hipMemsetAsync(segsum,  0, (size_t)n_ent * 2 * sizeof(float), stream);
        hipMemsetAsync(ent_agg, 0, (size_t)n_ent * C * sizeof(float), stream);
        hipMemsetAsync(usr_agg, 0, (size_t)n_users * C * sizeof(float), stream);

        // Q = ent_cur @ W_Q  (WMMA; one wave per 16 rows, 8 waves per block)
        int tiles = (n_ent + 15) / 16;
        int gq = (tiles + 7) / 8;
        qmat_kernel<<<gq, 256, 0, stream>>>(ent_cur, W_Q, Q, n_ent);

        score_kernel<<<(E + 255) / 256, 256, 0, stream>>>(
            (const float4*)Q, head, tail, edge_type,
            (const float4*)relation_emb, score, segmax, E);

        expsum_kernel<<<(2 * E + 255) / 256, 256, 0, stream>>>(
            score, head, segmax, segsum, E);

        long long na = (long long)E * 16;
        agg_kernel<<<(unsigned int)((na + 255) / 256), 256, 0, stream>>>(
            score, segsum, head, tail, edge_type,
            (const float4*)relation_emb, (const float4*)ent_cur, ent_agg, na);

        long long nu = (long long)NI * 16;
        user_agg_kernel<<<(unsigned int)((nu + 255) / 256), 256, 0, stream>>>(
            inter_w, u_idx, i_idx, (const float4*)ent_cur, usr_agg, nu);

        norm_res_kernel<<<(n_ent + 255) / 256, 256, 0, stream>>>(
            (const float4*)ent_agg, (float4*)ent_cur, (float4*)out_ent, n_ent);
        norm_res_kernel<<<(n_users + 255) / 256, 256, 0, stream>>>(
            (const float4*)usr_agg, (float4*)usr_cur, (float4*)out_usr, n_users);
    }
}
